// GPT2Attention_9440338117420
// MI455X (gfx1250) — compile-verified, matrix operand order checked by reading, not run on hardware
//
#include <hip/hip_runtime.h>
#include <hip/hip_bf16.h>

// ---------------------------------------------------------------------------
// GPT-2 causal self-attention forward for MI455X (gfx1250, wave32, WMMA+TDM).
//   out = proj( softmax_causal(QK^T/sqrt(Dh)) V ),  also returns K, V.
// f32 inputs are converted to f16 once; all matmuls run on
// v_wmma_f32_16x16x32_f16 with f32 accumulation. GEMM A-panels are staged
// into LDS (chunk-major, to spread LDS banks) with the Tensor Data Mover.
// ---------------------------------------------------------------------------

#define B_  4
#define T_  2048
#define C_  1024
#define H_  16
#define DH  64
#define M_  (B_ * T_)        // 8192 rows
#define N3  (3 * C_)         // 3072 qkv cols

typedef __attribute__((ext_vector_type(16))) _Float16 v16h;
typedef __attribute__((ext_vector_type(8)))  _Float16 v8h;
typedef __attribute__((ext_vector_type(4)))  _Float16 v4h;
typedef __attribute__((ext_vector_type(8)))  float    v8f;
typedef __attribute__((ext_vector_type(4)))  float    v4f;
typedef __attribute__((ext_vector_type(4)))  unsigned int v4u;
typedef __attribute__((ext_vector_type(8)))  int      v8i;
typedef __attribute__((ext_vector_type(4)))  int      v4i;

union V16H { v16h v; v8h h[2]; };

#ifndef __has_builtin
#define __has_builtin(x) 0
#endif
#if defined(__HIP_DEVICE_COMPILE__) && __has_builtin(__builtin_amdgcn_tensor_load_to_lds)
#define HAVE_TDM 1
#else
#define HAVE_TDM 0
#endif

// WMMA fragment loads (wave32 layouts, ISA 7.12.2):
// A (16x32 f16): lane L (row L&15): halves 0..7 = K[kb + hi*8 ..], 8..15 = K[kb+16+hi*8 ..]
// B (32x16 f16): same register pattern applied to B^T rows (row = lane&15).
__device__ __forceinline__ v16h load_ab_f16(const _Float16* __restrict__ row, int kb, int hi) {
    V16H u;
    const int o = kb + (hi ? 8 : 0);
    u.h[0] = *(const v8h*)(row + o);
    u.h[1] = *(const v8h*)(row + o + 16);
    return u.v;
}

__device__ __forceinline__ v8f wmma16(v16h a, v16h b, v8f c) {
    return __builtin_amdgcn_wmma_f32_16x16x32_f16(false, a, false, b, (short)0, c, false, false);
}

#if HAVE_TDM
// TDM: DMA a 32x1024-half A-panel (row stride C_ halves) from global memory
// into LDS at byte offset lds_off, stored CHUNK-MAJOR:
//   3-D tile x=32 halves (within chunk), y=32 rows, z=32 chunks,
//   global addr = base + 2*(x + y*1024 + z*32), LDS written contiguously
//   -> LDS half index = x + 32*y + 1024*z. D# per CDNA5 ISA.
__device__ __forceinline__ void tdm_load_panel(unsigned int lds_off, const void* g) {
    const unsigned long long ga = (unsigned long long)(uintptr_t)g;
    v4u g0;
    g0[0] = 1u;                                    // count=1 (valid descriptor)
    g0[1] = lds_off;                               // lds_addr [63:32]
    g0[2] = (unsigned int)ga;                      // global_addr[31:0]  -> bits 95:64
    g0[3] = (unsigned int)(ga >> 32) | (2u << 30); // global_addr[56:32] + type=2

    // group1: data_size=2B (code 1), no pad/iterate/multicast.
    unsigned long long u0 = (1ull << 16)        // data_size
                          | (32ull << 48);      // tensor_dim0 = 32 (bits 79:48 low part)
    unsigned long long u1 = (32ull << 16)       // tensor_dim1 = 32  (bits 111:80)
                          | (32ull << 48);      // tile_dim0  = 32  (bits 127:112)
    unsigned long long u2 = 32ull               // tile_dim1  = 32  (bits 143:128)
                          | (32ull << 16)       // tile_dim2  = 32  (bits 159:144)
                          | (1024ull << 32);    // tensor_dim0_stride[31:0] (bits 191:160)
    unsigned long long u3 = (32ull << 16);      // tensor_dim1_stride = 32 (bits 255:208)
    union { unsigned long long u[4]; v8i v; } g1;
    g1.u[0] = u0; g1.u[1] = u1; g1.u[2] = u2; g1.u[3] = u3;
    v4i g2 = {32, 0, 0, 0};                     // tensor_dim2 = 32
    v4i z4 = {0, 0, 0, 0};
#if __clang_major__ >= 23
    v8i z8 = {0, 0, 0, 0, 0, 0, 0, 0};
    __builtin_amdgcn_tensor_load_to_lds(g0, g1.v, g2, z4, z8, 0);
#else
    __builtin_amdgcn_tensor_load_to_lds(g0, g1.v, g2, z4, 0);
#endif
}
#endif

// Stage a 32-row A-panel into LDS (chunk-major). Must be called by all
// threads of the block; leaves panel visible after barrier. The asm escape
// forces the compiler to treat As as written (TDM stores are invisible to IR).
__device__ __forceinline__ void stage_panel(_Float16* As, const _Float16* gsrc) {
#if HAVE_TDM
    if ((threadIdx.x >> 5) == 0) {
        tdm_load_panel(0u, gsrc);
        __builtin_amdgcn_s_wait_tensorcnt(0);
    }
    asm volatile("" : : "r"(As) : "memory");   // As is written by TDM
    __syncthreads();
#else
    for (int i = threadIdx.x; i < 32 * C_ / 8; i += 256) {
        const int z = i >> 7;                  // chunk
        const int r = (i >> 2) & 31;           // row
        const int xg = i & 3;                  // 8-half group within chunk row
        ((v8h*)As)[(z << 7) + (r << 2) + xg] =
            *(const v8h*)(gsrc + (size_t)r * C_ + z * 32 + xg * 8);
    }
    __syncthreads();
#endif
}

// ---------------------------------------------------------------------------
// Stage 0: f32 -> f16 conversion (x, W_attn, W_proj), 4 elements/thread.
// ---------------------------------------------------------------------------
__global__ __launch_bounds__(256) void cvt_f16_kernel(const float* __restrict__ src,
                                                      _Float16* __restrict__ dst, int n4) {
    const int i = blockIdx.x * 256 + threadIdx.x;
    if (i < n4) {
        v4f s = ((const v4f*)src)[i];
        v4h d;
#pragma unroll
        for (int j = 0; j < 4; ++j) d[j] = (_Float16)s[j];
        ((v4h*)dst)[i] = d;
    }
}

// ---------------------------------------------------------------------------
// Stage 1: qkv = x @ W_attn^T + b_attn.
// Block: 8 waves share a TDM-staged 32x1024 f16 A-panel in LDS; each wave
// computes a 32x64 tile (8 WMMA accumulators). 256 m-tiles x 6 n-blocks.
//   q -> qh (f16); k -> d_out K region (f32) + kh (f16)
//   v -> d_out V region (f32) + vth (f16, transposed [B,H,Dh,T])
// ---------------------------------------------------------------------------
__global__ __launch_bounds__(256) void qkv_kernel(
    const _Float16* __restrict__ xh, const _Float16* __restrict__ Wah,
    const float* __restrict__ ba,
    float* __restrict__ outK, float* __restrict__ outV,
    _Float16* __restrict__ qh, _Float16* __restrict__ kh, _Float16* __restrict__ vth)
{
    __shared__ __align__(16) _Float16 As[32 * C_];   // 64 KB, chunk-major
    const int lane = threadIdx.x & 31;
    const int hi = lane >> 4, ln = lane & 15;
    const int warp = threadIdx.x >> 5;
    const int m0 = (blockIdx.x / 6) * 32;
    const int n0 = (blockIdx.x % 6) * 512 + warp * 64;

    stage_panel(As, xh + (size_t)m0 * C_);

    const _Float16* wrow[4];
#pragma unroll
    for (int nt = 0; nt < 4; ++nt)
        wrow[nt] = Wah + (size_t)(n0 + nt * 16 + ln) * C_;

    v8f c0[4], c1[4];
#pragma unroll
    for (int nt = 0; nt < 4; ++nt) { c0[nt] = (v8f)(0.0f); c1[nt] = (v8f)(0.0f); }

    for (int k = 0; k < C_; k += 32) {
        const _Float16* ap = As + (k >> 5) * 1024;   // chunk block: 32 rows x 32 halves
        v16h a0 = load_ab_f16(ap + ln * 32, 0, hi);
        v16h a1 = load_ab_f16(ap + (16 + ln) * 32, 0, hi);
        v16h b[4];
#pragma unroll
        for (int nt = 0; nt < 4; ++nt) b[nt] = load_ab_f16(wrow[nt], k, hi);
#pragma unroll
        for (int nt = 0; nt < 4; ++nt) {
            c0[nt] = wmma16(a0, b[nt], c0[nt]);
            c1[nt] = wmma16(a1, b[nt], c1[nt]);
        }
    }

#pragma unroll
    for (int ms = 0; ms < 2; ++ms) {
#pragma unroll
        for (int nt = 0; nt < 4; ++nt) {
            const int n = n0 + nt * 16 + ln;
            const int g = n >> 10;               // 0:q 1:k 2:v
            const int cc = n & (C_ - 1);
            const int h = cc >> 6, d = cc & 63;
            const float bias = ba[n];
            const v8f acc = ms ? c1[nt] : c0[nt];
#pragma unroll
            for (int j = 0; j < 8; ++j) {
                const int m = m0 + ms * 16 + j + hi * 8;
                const int b = m >> 11, t = m & (T_ - 1);
                const size_t idx = (((size_t)(b * H_ + h)) * T_ + t) * DH + d;
                const float v = acc[j] + bias;
                if (g == 0) {
                    qh[idx] = (_Float16)v;
                } else if (g == 1) {
                    outK[idx] = v;
                    kh[idx] = (_Float16)v;
                } else {
                    outV[idx] = v;
                    vth[(((size_t)(b * H_ + h)) * DH + d) * T_ + t] = (_Float16)v;
                }
            }
        }
    }
}

// ---------------------------------------------------------------------------
// Stage 2: flash attention. One wave owns 16 query rows of one (b,h).
// Online softmax; probs transposed C-layout -> A-layout through LDS.
// ---------------------------------------------------------------------------
__global__ __launch_bounds__(256) void attn_kernel(
    const _Float16* __restrict__ qh, const _Float16* __restrict__ kh,
    const _Float16* __restrict__ vth, _Float16* __restrict__ yh)
{
    __shared__ __align__(16) _Float16 lds[8][16 * 32];  // per-wave 16x32 probs
    const int lane = threadIdx.x & 31;
    const int hi = lane >> 4, ln = lane & 15;
    const int warp = threadIdx.x >> 5;
    const int task = blockIdx.x * 8 + warp;
    const int bh = task >> 7;                // T_/16 = 128 q-tiles per head
    const int q0 = (task & 127) * 16;
    const float scale = 0.125f;              // 1/sqrt(64)

    const _Float16* qrow = qh + ((size_t)bh * T_ + q0 + ln) * DH;
    const v16h qa0 = load_ab_f16(qrow, 0, hi);
    const v16h qa1 = load_ab_f16(qrow, 32, hi);

    v8f o[4];
#pragma unroll
    for (int nt = 0; nt < 4; ++nt) o[nt] = (v8f)(0.0f);
    float mrow[8], lrow[8];
#pragma unroll
    for (int j = 0; j < 8; ++j) { mrow[j] = -1.0e30f; lrow[j] = 0.0f; }

    _Float16* P = &lds[warp][0];

    for (int kb = 0; kb < q0 + 16; kb += 32) {
        // ---- S = Q K^T for two 16-wide key tiles (contract over Dh=64) ----
        v8f s[2];
#pragma unroll
        for (int t = 0; t < 2; ++t) {
            const _Float16* krow = kh + ((size_t)bh * T_ + kb + t * 16 + ln) * DH;
            v16h b0 = load_ab_f16(krow, 0, hi);
            v16h b1 = load_ab_f16(krow, 32, hi);
            v8f acc = (v8f)(0.0f);
            acc = wmma16(qa0, b0, acc);
            acc = wmma16(qa1, b1, acc);
            s[t] = acc;
        }
        // ---- causal mask + online softmax (rows split across lane halves) ----
#pragma unroll
        for (int j = 0; j < 8; ++j) {
            const int q = q0 + j + hi * 8;
            const float v0 = (kb + ln      <= q) ? s[0][j] * scale : -1.0e30f;
            const float v1 = (kb + 16 + ln <= q) ? s[1][j] * scale : -1.0e30f;
            float mx = fmaxf(v0, v1);
#pragma unroll
            for (int d = 1; d < 16; d <<= 1) mx = fmaxf(mx, __shfl_xor(mx, d, 32));
            const float mnew = fmaxf(mrow[j], mx);
            const float alpha = __expf(mrow[j] - mnew);
            const float p0 = __expf(v0 - mnew);
            const float p1 = __expf(v1 - mnew);
            float rs = p0 + p1;
#pragma unroll
            for (int d = 1; d < 16; d <<= 1) rs += __shfl_xor(rs, d, 32);
            lrow[j] = lrow[j] * alpha + rs;
            mrow[j] = mnew;
#pragma unroll
            for (int nt = 0; nt < 4; ++nt) o[nt][j] *= alpha;
            const int r = j + hi * 8;
            P[r * 32 + ln]      = (_Float16)p0;
            P[r * 32 + 16 + ln] = (_Float16)p1;
        }
        asm volatile("s_wait_dscnt 0" ::: "memory");   // same-wave LDS RAW fence
        __builtin_amdgcn_wave_barrier();
        // ---- re-read probs in A-operand layout, then O += P V ----
        const v16h pa = load_ab_f16(P + ln * 32, 0, hi);
#pragma unroll
        for (int nt = 0; nt < 4; ++nt) {
            const _Float16* vrow = vth + ((size_t)bh * DH + nt * 16 + ln) * T_ + kb;
            v16h vb = load_ab_f16(vrow, 0, hi);
            o[nt] = wmma16(pa, vb, o[nt]);
        }
    }

    // ---- epilogue: y[b,t,h*64+d] = o / l ----
    const int b = bh >> 4, h = bh & 15;
#pragma unroll
    for (int nt = 0; nt < 4; ++nt) {
        const int d = nt * 16 + ln;
#pragma unroll
        for (int j = 0; j < 8; ++j) {
            const int t = q0 + j + hi * 8;
            const float v = o[nt][j] / lrow[j];
            yh[((size_t)(b * T_ + t)) * C_ + h * DH + d] = (_Float16)v;
        }
    }
}

// ---------------------------------------------------------------------------
// Stage 3: out = y @ W_proj^T + b_proj. Same block structure as stage 1:
// TDM-staged 32x1024 A-panel, 32x64 tile per wave. 256 m-tiles x 2 n-blocks.
// ---------------------------------------------------------------------------
__global__ __launch_bounds__(256) void proj_kernel(
    const _Float16* __restrict__ yh, const _Float16* __restrict__ Wph,
    const float* __restrict__ bp, float* __restrict__ out)
{
    __shared__ __align__(16) _Float16 As[32 * C_];   // 64 KB, chunk-major
    const int lane = threadIdx.x & 31;
    const int hi = lane >> 4, ln = lane & 15;
    const int warp = threadIdx.x >> 5;
    const int m0 = (blockIdx.x / 2) * 32;
    const int n0 = (blockIdx.x % 2) * 512 + warp * 64;

    stage_panel(As, yh + (size_t)m0 * C_);

    const _Float16* wrow[4];
#pragma unroll
    for (int nt = 0; nt < 4; ++nt)
        wrow[nt] = Wph + (size_t)(n0 + nt * 16 + ln) * C_;

    v8f c0[4], c1[4];
#pragma unroll
    for (int nt = 0; nt < 4; ++nt) { c0[nt] = (v8f)(0.0f); c1[nt] = (v8f)(0.0f); }

    for (int k = 0; k < C_; k += 32) {
        const _Float16* ap = As + (k >> 5) * 1024;
        v16h a0 = load_ab_f16(ap + ln * 32, 0, hi);
        v16h a1 = load_ab_f16(ap + (16 + ln) * 32, 0, hi);
        v16h b[4];
#pragma unroll
        for (int nt = 0; nt < 4; ++nt) b[nt] = load_ab_f16(wrow[nt], k, hi);
#pragma unroll
        for (int nt = 0; nt < 4; ++nt) {
            c0[nt] = wmma16(a0, b[nt], c0[nt]);
            c1[nt] = wmma16(a1, b[nt], c1[nt]);
        }
    }

#pragma unroll
    for (int ms = 0; ms < 2; ++ms) {
#pragma unroll
        for (int nt = 0; nt < 4; ++nt) {
            const int n = n0 + nt * 16 + ln;
            const float bias = bp[n];
            const v8f acc = ms ? c1[nt] : c0[nt];
#pragma unroll
            for (int j = 0; j < 8; ++j) {
                const int m = m0 + ms * 16 + j + hi * 8;
                out[(size_t)m * C_ + n] = acc[j] + bias;
            }
        }
    }
}

// ---------------------------------------------------------------------------
extern "C" void kernel_launch(void* const* d_in, const int* in_sizes, int n_in,
                              void* d_out, int out_size, void* d_ws, size_t ws_size,
                              hipStream_t stream) {
    const float* x  = (const float*)d_in[0];
    const float* Wa = (const float*)d_in[1];
    const float* ba = (const float*)d_in[2];
    const float* Wp = (const float*)d_in[3];
    const float* bp = (const float*)d_in[4];

    float* out = (float*)d_out;
    const size_t KV = (size_t)B_ * H_ * T_ * DH;     // 8,388,608
    float* outK = out + KV;
    float* outV = out + 2 * KV;

    _Float16* qh  = (_Float16*)d_ws;                 // [B,H,T,Dh] f16
    _Float16* kh  = qh  + KV;                        // [B,H,T,Dh] f16
    _Float16* vth = kh  + KV;                        // [B,H,Dh,T] f16 (V^T)
    _Float16* yh  = vth + KV;                        // [B,T,C]    f16
    _Float16* xh  = yh  + KV;                        // [B,T,C]    f16
    _Float16* Wah = xh  + (size_t)M_ * C_;           // [3C,C]     f16
    _Float16* Wph = Wah + (size_t)N3 * C_;           // [C,C]      f16
    (void)in_sizes; (void)n_in; (void)out_size; (void)ws_size;

    // Stage 0: one-time f32 -> f16 conversions.
    cvt_f16_kernel<<<dim3((M_ * C_ / 4) / 256), dim3(256), 0, stream>>>(x, xh, M_ * C_ / 4);
    cvt_f16_kernel<<<dim3((N3 * C_ / 4) / 256), dim3(256), 0, stream>>>(Wa, Wah, N3 * C_ / 4);
    cvt_f16_kernel<<<dim3((C_ * C_ / 4) / 256), dim3(256), 0, stream>>>(Wp, Wph, C_ * C_ / 4);

    // Stage 1: 256 m-tiles x 6 n-blocks.
    qkv_kernel<<<dim3((M_ / 32) * (N3 / 512)), dim3(256), 0, stream>>>(
        xh, Wah, ba, outK, outV, qh, kh, vth);
    // Stage 2: B*H * (T/16) q-tiles / 8 waves per block.
    attn_kernel<<<dim3(B_ * H_ * (T_ / 16) / 8), dim3(256), 0, stream>>>(
        qh, kh, vth, yh);
    // Stage 3: 256 m-tiles x 2 n-blocks.
    proj_kernel<<<dim3((M_ / 32) * (C_ / 512)), dim3(256), 0, stream>>>(
        yh, Wph, bp, out);
}
